// Block_45552423141629
// MI455X (gfx1250) — matrix operand order repaired and checked by reading, not compiled, not run on hardware
//
#include <hip/hip_runtime.h>
#include <math.h>

// ---------------------------------------------------------------------------
// Problem constants (match reference)
// ---------------------------------------------------------------------------
#define BSZ   4
#define SEQ   8192
#define DIM   1024
#define NH    16
#define HD    64
#define DFF   4096
#define TOK   (BSZ * SEQ)       // 32768 tokens
#define EPS_LN   1e-5f
#define EPS_NORM 1e-6f

typedef __attribute__((ext_vector_type(16))) __bf16 v16bf;
typedef __attribute__((ext_vector_type(8)))  __bf16 v8bf;
typedef __attribute__((ext_vector_type(4)))  __bf16 v4bf;
typedef __attribute__((ext_vector_type(2)))  __bf16 v2bf;
typedef __attribute__((ext_vector_type(8)))  float  v8f;

// ---------------------------------------------------------------------------
// WMMA helpers (CDNA5: V_WMMA_F32_16X16X32_BF16, wave32)
// ---------------------------------------------------------------------------
__device__ __forceinline__ v8f wmma_bf16(v16bf a, v16bf b, v8f c) {
    return __builtin_amdgcn_wmma_f32_16x16x32_bf16(
        /*neg_a=*/false, a, /*neg_b=*/false, b,
        /*c_mod=*/(short)0, c, /*reuse_a=*/false, /*reuse_b=*/false);
}

// A-matrix 16x32 bf16 fragment (ISA 7.12.2):
//  lanes 0-15 : VGPR0..3 -> K=0..7,  VGPR4..7 -> K=16..23   (half=0)
//  lanes 16-31: VGPR0..3 -> K=8..15, VGPR4..7 -> K=24..31   (half=1)
// p points at row m of the tile (row-major, K contiguous).
// (vectorizes to 2x ds_load_b128 / global_load_b128)
__device__ __forceinline__ v16bf frag_a(const __bf16* p, int half) {
    v16bf f;
    int o1 = 8 * half;
    int o2 = 16 + 8 * half;
#pragma unroll
    for (int j = 0; j < 4; ++j) {
        v2bf t = *(const v2bf*)(p + o1 + 2 * j);
        f[2 * j] = t[0]; f[2 * j + 1] = t[1];
    }
#pragma unroll
    for (int j = 0; j < 4; ++j) {
        v2bf t = *(const v2bf*)(p + o2 + 2 * j);
        f[8 + 2 * j] = t[0]; f[8 + 2 * j + 1] = t[1];
    }
    return f;
}

// B-matrix 32x16 bf16 fragment: VGPR0..3 -> K=8*half+0..7, VGPR4..7 -> K=16+8*half+0..7 ; column = lane%16.
// p points at column n's data laid out K-contiguous (i.e. Bt row n).
__device__ __forceinline__ v16bf frag_b(const __bf16* p, int half) {
    v16bf f;
    int o = 8 * half;
#pragma unroll
    for (int j = 0; j < 8; ++j) {
        v2bf t = *(const v2bf*)(p + o + (j >> 2) * 16 + 2 * (j & 3));
        f[2 * j] = t[0]; f[2 * j + 1] = t[1];
    }
    return f;
}

// ---------------------------------------------------------------------------
// Weight convert: f32 W[K][N] -> bf16 Wt[N][K]
// ---------------------------------------------------------------------------
__global__ void transpose_w_kernel(const float* __restrict__ in,
                                   __bf16* __restrict__ out, int K, int N) {
    size_t i = (size_t)blockIdx.x * 256 + threadIdx.x;
    if (i >= (size_t)K * N) return;
    int k = (int)(i / N);
    int n = (int)(i % N);
    out[(size_t)n * K + k] = (__bf16)in[i];
}

// ---------------------------------------------------------------------------
// Zero-fill f32
// ---------------------------------------------------------------------------
__global__ void zero_f32_kernel(float* __restrict__ p, int n) {
    int i = blockIdx.x * 256 + threadIdx.x;
    if (i < n) p[i] = 0.f;
}

// ---------------------------------------------------------------------------
// LayerNorm: one block per token row of 1024 f32 -> bf16
// ---------------------------------------------------------------------------
__global__ __launch_bounds__(256)
void ln_kernel(const float* __restrict__ x, const float* __restrict__ w,
               const float* __restrict__ b, __bf16* __restrict__ out) {
    int row = blockIdx.x, tid = threadIdx.x;
    const float4 v = ((const float4*)(x + (size_t)row * DIM))[tid];
    float s = v.x + v.y + v.z + v.w;
    float q = v.x * v.x + v.y * v.y + v.z * v.z + v.w * v.w;
#pragma unroll
    for (int o = 16; o > 0; o >>= 1) {
        s += __shfl_down(s, o, 32);
        q += __shfl_down(q, o, 32);
    }
    __shared__ float sh[16];
    int lane = tid & 31, wv = tid >> 5;
    if (lane == 0) { sh[wv] = s; sh[8 + wv] = q; }
    __syncthreads();
    if (tid == 0) {
        float ts = 0.f, tq = 0.f;
#pragma unroll
        for (int i = 0; i < 8; ++i) { ts += sh[i]; tq += sh[8 + i]; }
        sh[0] = ts; sh[8] = tq;
    }
    __syncthreads();
    float mu = sh[0] * (1.f / DIM);
    float var = sh[8] * (1.f / DIM) - mu * mu;
    float rs = rsqrtf(var + EPS_LN);
    int d = tid * 4;
    v4bf o4;
    o4[0] = (__bf16)((v.x - mu) * rs * w[d + 0] + b[d + 0]);
    o4[1] = (__bf16)((v.y - mu) * rs * w[d + 1] + b[d + 1]);
    o4[2] = (__bf16)((v.z - mu) * rs * w[d + 2] + b[d + 2]);
    o4[3] = (__bf16)((v.w - mu) * rs * w[d + 3] + b[d + 3]);
    *(v4bf*)&out[(size_t)row * DIM + d] = o4;
}

// ---------------------------------------------------------------------------
// Generic bf16 WMMA GEMM: C[M,N] = A[M,K] * Bt[N,K]^T, fused epilogues.
// Block tile 128x128, BK=32, 256 threads = 8 waves (4x2), wave tile 32x64.
// LDS double-buffered: global loads for slab k+1 overlap WMMA on slab k,
// one barrier per K-step.
// EPI: 0 raw bf16, 1 phi(elu+1) bf16, 2 bias+GELU bf16,
//      3 +resid f32, 4 bias+resid f32
// ---------------------------------------------------------------------------
template <int EPI>
__global__ __launch_bounds__(256)
void gemm_kernel(const __bf16* __restrict__ A, const __bf16* __restrict__ Bt,
                 const float* __restrict__ bias, const float* __restrict__ resid,
                 void* __restrict__ outp, int M, int N, int K) {
    __shared__ __bf16 As[2][128][40];
    __shared__ __bf16 Bs[2][128][40];
    const int tid = threadIdx.x;
    const int lane = tid & 31, wave = tid >> 5;
    const int wm = wave >> 1, wn = wave & 1;
    const int half = lane >> 4, l16 = lane & 15;
    const int m0 = blockIdx.x * 128, n0 = blockIdx.y * 128;

    // per-thread staging slots: 512 v8bf chunks per 128x32 tile
    const int u0 = tid, u1 = tid + 256;
    const int rowA0 = u0 >> 2, colA0 = u0 & 3;
    const int rowA1 = u1 >> 2, colA1 = u1 & 3;

    v8f acc[2][4];
#pragma unroll
    for (int mt = 0; mt < 2; ++mt)
#pragma unroll
        for (int nt = 0; nt < 4; ++nt)
#pragma unroll
            for (int r = 0; r < 8; ++r) acc[mt][nt][r] = 0.f;

    // prologue: stage slab 0 into buffer 0
    v8bf ra0 = *(const v8bf*)&A[(size_t)(m0 + rowA0) * K + colA0 * 8];
    v8bf ra1 = *(const v8bf*)&A[(size_t)(m0 + rowA1) * K + colA1 * 8];
    v8bf rb0 = *(const v8bf*)&Bt[(size_t)(n0 + rowA0) * K + colA0 * 8];
    v8bf rb1 = *(const v8bf*)&Bt[(size_t)(n0 + rowA1) * K + colA1 * 8];
    *(v8bf*)&As[0][rowA0][colA0 * 8] = ra0;
    *(v8bf*)&As[0][rowA1][colA1 * 8] = ra1;
    *(v8bf*)&Bs[0][rowA0][colA0 * 8] = rb0;
    *(v8bf*)&Bs[0][rowA1][colA1 * 8] = rb1;
    __syncthreads();

    int cur = 0;
    for (int k0 = 0; k0 < K; k0 += 32) {
        const bool more = (k0 + 32) < K;
        if (more) {   // issue next slab's global loads early (latency overlap)
            int kn = k0 + 32;
            ra0 = *(const v8bf*)&A[(size_t)(m0 + rowA0) * K + kn + colA0 * 8];
            ra1 = *(const v8bf*)&A[(size_t)(m0 + rowA1) * K + kn + colA1 * 8];
            rb0 = *(const v8bf*)&Bt[(size_t)(n0 + rowA0) * K + kn + colA0 * 8];
            rb1 = *(const v8bf*)&Bt[(size_t)(n0 + rowA1) * K + kn + colA1 * 8];
        }
        v16bf af[2], bfr[4];
#pragma unroll
        for (int mt = 0; mt < 2; ++mt)
            af[mt] = frag_a(&As[cur][wm * 32 + mt * 16 + l16][0], half);
#pragma unroll
        for (int nt = 0; nt < 4; ++nt)
            bfr[nt] = frag_b(&Bs[cur][wn * 64 + nt * 16 + l16][0], half);
#pragma unroll
        for (int mt = 0; mt < 2; ++mt)
#pragma unroll
            for (int nt = 0; nt < 4; ++nt)
                acc[mt][nt] = wmma_bf16(af[mt], bfr[nt], acc[mt][nt]);
        if (more) {   // write the *other* buffer; one barrier per iteration
            int nxt = cur ^ 1;
            *(v8bf*)&As[nxt][rowA0][colA0 * 8] = ra0;
            *(v8bf*)&As[nxt][rowA1][colA1 * 8] = ra1;
            *(v8bf*)&Bs[nxt][rowA0][colA0 * 8] = rb0;
            *(v8bf*)&Bs[nxt][rowA1][colA1 * 8] = rb1;
            __syncthreads();
            cur = nxt;
        }
    }

#pragma unroll
    for (int mt = 0; mt < 2; ++mt) {
#pragma unroll
        for (int nt = 0; nt < 4; ++nt) {
            int gn = n0 + wn * 64 + nt * 16 + l16;
#pragma unroll
            for (int r = 0; r < 8; ++r) {
                int gm = m0 + wm * 32 + mt * 16 + r + 8 * half;
                size_t idx = (size_t)gm * N + gn;
                float x = acc[mt][nt][r];
                if (EPI == 0) {
                    ((__bf16*)outp)[idx] = (__bf16)x;
                } else if (EPI == 1) {             // phi = elu(x)+1
                    float y = x > 0.f ? x + 1.f : expf(x);
                    ((__bf16*)outp)[idx] = (__bf16)y;
                } else if (EPI == 2) {             // exact GELU with bias
                    float y = x + bias[gn];
                    float g = 0.5f * y * (1.f + erff(y * 0.70710678118654752f));
                    ((__bf16*)outp)[idx] = (__bf16)g;
                } else if (EPI == 3) {
                    ((float*)outp)[idx] = x + resid[idx];
                } else {                           // EPI == 4
                    ((float*)outp)[idx] = x + bias[gn] + resid[idx];
                }
            }
        }
    }
}

// ---------------------------------------------------------------------------
// Per-head partial kv = k^T v over an s-chunk, accumulated into f32 KVF via
// native global_atomic_add_f32. KVF layout: [bh][e(0..79)][d(0..63)],
// e 0-63 = kv columns (transposed), e 64 = ksum, e 65-79 zero-padded later.
// grid (B*H, SCHUNKS); 8 waves: wave = (et-pair<<2) | dt.
// ---------------------------------------------------------------------------
#define SCHUNKS 8
__global__ __launch_bounds__(256)
void kv_part_kernel(const __bf16* __restrict__ Kp, const __bf16* __restrict__ Vp,
                    float* __restrict__ KVF) {
    __shared__ __bf16 kT[64][40];
    __shared__ __bf16 vT[64][40];
    const int bh = blockIdx.x;
    const int b = bh >> 4, h = bh & 15;
    const int tid = threadIdx.x, lane = tid & 31, wave = tid >> 5;
    const int l16 = lane & 15, half = lane >> 4;
    const int dt = wave & 3, ep = wave >> 2;
    float* out = KVF + (size_t)bh * (80 * 64);

    v8f acc0, acc1;
#pragma unroll
    for (int r = 0; r < 8; ++r) { acc0[r] = 0.f; acc1[r] = 0.f; }
    float ks = 0.f;

    const size_t base = ((size_t)b * SEQ) * DIM + h * HD;
    const int sld = tid >> 3, dg = tid & 7;
    const int sbeg = blockIdx.y * (SEQ / SCHUNKS);
    const int send = sbeg + (SEQ / SCHUNKS);

    for (int s0 = sbeg; s0 < send; s0 += 32) {
        size_t g = base + (size_t)(s0 + sld) * DIM + dg * 8;
        v8bf k8 = *(const v8bf*)&Kp[g];
        v8bf v8 = *(const v8bf*)&Vp[g];
        __syncthreads();  // previous iteration's frag reads done
#pragma unroll
        for (int i = 0; i < 8; ++i) {
            kT[dg * 8 + i][sld] = k8[i];
            vT[dg * 8 + i][sld] = v8[i];
        }
        __syncthreads();
        v16bf a  = frag_a(&kT[dt * 16 + l16][0], half);
        v16bf b0 = frag_b(&vT[(2 * ep + 0) * 16 + l16][0], half);
        v16bf b1 = frag_b(&vT[(2 * ep + 1) * 16 + l16][0], half);
        acc0 = wmma_bf16(a, b0, acc0);
        acc1 = wmma_bf16(a, b1, acc1);
        if (ep == 0) {
            float p = 0.f;
#pragma unroll
            for (int e = 0; e < 16; ++e) p += (float)a[e];
            ks += p;
        }
    }

    // accumulate partial tiles (transposed): out[e*64 + d]
#pragma unroll
    for (int r = 0; r < 8; ++r) {
        int d = dt * 16 + r + 8 * half;
        unsafeAtomicAdd(&out[((2 * ep + 0) * 16 + l16) * 64 + d], acc0[r]);
        unsafeAtomicAdd(&out[((2 * ep + 1) * 16 + l16) * 64 + d], acc1[r]);
    }
    if (ep == 0) {
        float tot = ks + __shfl_xor(ks, 16, 32);
        if (half == 0)
            unsafeAtomicAdd(&out[64 * 64 + dt * 16 + l16], tot);
    }
}

// ---------------------------------------------------------------------------
// Pack KVF f32 -> kvxT bf16 (zero rows e=65..79)
// ---------------------------------------------------------------------------
__global__ void kv_pack_kernel(const float* __restrict__ KVF,
                               __bf16* __restrict__ kvxT) {
    int i = blockIdx.x * 256 + threadIdx.x;      // over 64*80*64
    int e = (i >> 6) % 80;
    kvxT[i] = (e <= 64) ? (__bf16)KVF[i] : (__bf16)0.f;
}

// ---------------------------------------------------------------------------
// attn: out = (q @ kv) / (q . ksum + eps), per head; q has phi applied.
// grid (SEQ/256, B*H); per wave: 32 rows x 5 n-tiles (5th = normalizer col).
// ---------------------------------------------------------------------------
__global__ __launch_bounds__(256)
void attn_kernel(const __bf16* __restrict__ Q, const __bf16* __restrict__ kvxT,
                 __bf16* __restrict__ ATT) {
    __shared__ __bf16 kvs[80][68];    // padded: conflict-free frag_b reads
    const int bh = blockIdx.y;
    const int b = bh >> 4, h = bh & 15;
    const int tid = threadIdx.x, lane = tid & 31, wave = tid >> 5;
    const int l16 = lane & 15, half = lane >> 4;

    const __bf16* src = kvxT + (size_t)bh * (80 * 64);
    for (int i = tid; i < 80 * 64; i += 256) kvs[i / 64][i % 64] = src[i];
    __syncthreads();

    const int s0 = blockIdx.x * 256 + wave * 32;
    const size_t qbase = ((size_t)b * SEQ + s0) * DIM + h * HD;

    v8f acc[2][5];
#pragma unroll
    for (int mt = 0; mt < 2; ++mt)
#pragma unroll
        for (int nt = 0; nt < 5; ++nt)
#pragma unroll
            for (int r = 0; r < 8; ++r) acc[mt][nt][r] = 0.f;

#pragma unroll
    for (int kk = 0; kk < 2; ++kk) {   // K = 64 = 2 x 32
        v16bf af[2];
#pragma unroll
        for (int mt = 0; mt < 2; ++mt)
            af[mt] = frag_a(&Q[qbase + (size_t)(mt * 16 + l16) * DIM + kk * 32], half);
#pragma unroll
        for (int nt = 0; nt < 5; ++nt) {
            v16bf bfr = frag_b(&kvs[nt * 16 + l16][kk * 32], half);
#pragma unroll
            for (int mt = 0; mt < 2; ++mt)
                acc[mt][nt] = wmma_bf16(af[mt], bfr, acc[mt][nt]);
        }
    }

#pragma unroll
    for (int mt = 0; mt < 2; ++mt) {
#pragma unroll
        for (int r = 0; r < 8; ++r) {
            // normalizer lives in tile 4, column 0 -> lane 16*half, element r
            float nv = __shfl(acc[mt][4][r], lane & 16, 32);
            float inv = 1.f / (nv + EPS_NORM);
            int m = s0 + mt * 16 + r + 8 * half;
            size_t ob = ((size_t)b * SEQ + m) * DIM + h * HD;
#pragma unroll
            for (int nt = 0; nt < 4; ++nt)
                ATT[ob + nt * 16 + l16] = (__bf16)(acc[mt][nt][r] * inv);
        }
    }
}

// ---------------------------------------------------------------------------
// Host-side orchestration
// ---------------------------------------------------------------------------
extern "C" void kernel_launch(void* const* d_in, const int* in_sizes, int n_in,
                              void* d_out, int out_size, void* d_ws, size_t ws_size,
                              hipStream_t stream) {
    (void)in_sizes; (void)n_in; (void)out_size; (void)ws_size;
    const float* src   = (const float*)d_in[0];
    const float* ln1_w = (const float*)d_in[1];
    const float* ln1_b = (const float*)d_in[2];
    const float* wq    = (const float*)d_in[3];
    const float* wk    = (const float*)d_in[4];
    const float* wv    = (const float*)d_in[5];
    const float* wo    = (const float*)d_in[6];
    const float* ln2_w = (const float*)d_in[7];
    const float* ln2_b = (const float*)d_in[8];
    const float* fc1_w = (const float*)d_in[9];
    const float* fc1_b = (const float*)d_in[10];
    const float* fc2_w = (const float*)d_in[11];
    const float* fc2_b = (const float*)d_in[12];

    char* ws = (char*)d_ws;
    size_t off = 0;
    auto alloc = [&](size_t bytes) -> void* {
        void* p = ws + off;
        off += (bytes + 255) & ~(size_t)255;
        return p;
    };
    const size_t wsz  = (size_t)DIM * DIM * 2;
    const size_t fsz  = (size_t)DIM * DFF * 2;
    const size_t bsz  = (size_t)TOK * DIM * 2;   // 64 MB bf16 activation

    __bf16* wqT  = (__bf16*)alloc(wsz);
    __bf16* wkT  = (__bf16*)alloc(wsz);
    __bf16* wvT  = (__bf16*)alloc(wsz);
    __bf16* woT  = (__bf16*)alloc(wsz);
    __bf16* f1T  = (__bf16*)alloc(fsz);          // [DFF][DIM]
    __bf16* f2T  = (__bf16*)alloc(fsz);          // [DIM][DFF]
    __bf16* kvxT = (__bf16*)alloc((size_t)BSZ * NH * 80 * 64 * 2);
    float*  KVF  = (float*)alloc((size_t)BSZ * NH * 80 * 64 * 4);
    __bf16* S0   = (__bf16*)alloc(bsz);          // h2 (LN2 out)
    __bf16* S1   = (__bf16*)alloc(bsz);          // q   / later G start
    __bf16* S2   = (__bf16*)alloc(bsz);          // k   / later attn
    __bf16* S3   = (__bf16*)alloc(bsz);          // v
    __bf16* S4   = (__bf16*)alloc(bsz);          // h1 (LN1 out)
    float*  R    = (float*)alloc((size_t)TOK * DIM * 4);  // post-attn residual

    // 1. weight convert + transpose (f32 -> bf16 Wt[N][K]); zero kv accum
    {
        int g1 = (DIM * DIM + 255) / 256;
        transpose_w_kernel<<<g1, 256, 0, stream>>>(wq, wqT, DIM, DIM);
        transpose_w_kernel<<<g1, 256, 0, stream>>>(wk, wkT, DIM, DIM);
        transpose_w_kernel<<<g1, 256, 0, stream>>>(wv, wvT, DIM, DIM);
        transpose_w_kernel<<<g1, 256, 0, stream>>>(wo, woT, DIM, DIM);
        int g2 = (DIM * DFF + 255) / 256;
        transpose_w_kernel<<<g2, 256, 0, stream>>>(fc1_w, f1T, DIM, DFF);
        transpose_w_kernel<<<g2, 256, 0, stream>>>(fc2_w, f2T, DFF, DIM);
        int nkv = BSZ * NH * 80 * 64;
        zero_f32_kernel<<<(nkv + 255) / 256, 256, 0, stream>>>(KVF, nkv);
    }

    // 2. h1 = LN1(src)
    ln_kernel<<<TOK, 256, 0, stream>>>(src, ln1_w, ln1_b, S4);

    // 3. q = phi(h1 wq), k = phi(h1 wk), v = h1 wv
    dim3 gQ(TOK / 128, DIM / 128);
    gemm_kernel<1><<<gQ, 256, 0, stream>>>(S4, wqT, nullptr, nullptr, S1, TOK, DIM, DIM);
    gemm_kernel<1><<<gQ, 256, 0, stream>>>(S4, wkT, nullptr, nullptr, S2, TOK, DIM, DIM);
    gemm_kernel<0><<<gQ, 256, 0, stream>>>(S4, wvT, nullptr, nullptr, S3, TOK, DIM, DIM);

    // 4. per-head kv = k^T v, ksum (chunked over s; f32 atomics), then pack
    kv_part_kernel<<<dim3(BSZ * NH, SCHUNKS), 256, 0, stream>>>(S2, S3, KVF);
    kv_pack_kernel<<<(BSZ * NH * 80 * 64) / 256, 256, 0, stream>>>(KVF, kvxT);

    // 5. attn = (q kv) / (q . ksum + eps)
    attn_kernel<<<dim3(SEQ / 256, BSZ * NH), 256, 0, stream>>>(S1, kvxT, S2);

    // 6. R = attn wo + src
    gemm_kernel<3><<<gQ, 256, 0, stream>>>(S2, woT, nullptr, src, R, TOK, DIM, DIM);

    // 7. h2 = LN2(R)
    ln_kernel<<<TOK, 256, 0, stream>>>(R, ln2_w, ln2_b, S0);

    // 8. G = gelu(h2 fc1 + b1)  (256 MB bf16 spanning S1..S4, all dead now)
    dim3 gF1(TOK / 128, DFF / 128);
    gemm_kernel<2><<<gF1, 256, 0, stream>>>(S0, f1T, fc1_b, nullptr, S1, TOK, DFF, DIM);

    // 9. out = G fc2 + b2 + R
    gemm_kernel<4><<<gQ, 256, 0, stream>>>(S1, f2T, fc2_b, R, (float*)d_out, TOK, DIM, DFF);
}
